// LinearMoCoreLoraLayer_85598698209648
// MI455X (gfx1250) — matrix operand order repaired and checked by reading, not compiled, not run on hardware
//
#include <hip/hip_runtime.h>
#include <stdint.h>

// Problem dims (fixed by reference)
#define B_DIM 4
#define S_DIM 2048
#define D_DIM 4096            // K of base GEMM
#define O_DIM 4096            // N of base GEMM
#define R_DIM 16
#define E_DIM 8
#define M_TOK (B_DIM * S_DIM) // 8192 rows
static constexpr float K_SCALING = 32.0f / 16.0f;

typedef __attribute__((ext_vector_type(16))) __bf16 v16bf;
typedef __attribute__((ext_vector_type(8)))  float  v8f;
typedef unsigned int u32x4 __attribute__((ext_vector_type(4)));
typedef int          i32x8 __attribute__((ext_vector_type(8)));
typedef int          i32x4 __attribute__((ext_vector_type(4)));

union FragAB { v16bf v; uint4 q[2]; };

__device__ __forceinline__ v8f wmma_bf16(v16bf a, v16bf b, v8f c) {
  // D = A*B + C, f32 accumulate; emits v_wmma_f32_16x16x32_bf16
  return __builtin_amdgcn_wmma_f32_16x16x32_bf16(false, a, false, b, (short)0, c, false, false);
}

// round-to-nearest-even f32 -> bf16 (as raw ushort)
__device__ __forceinline__ unsigned short f2bf(float f) {
  unsigned u = __float_as_uint(f);
  u += 0x7FFFu + ((u >> 16) & 1u);
  return (unsigned short)(u >> 16);
}
__device__ __forceinline__ float bf2f(unsigned short h) {
  return __uint_as_float(((unsigned)h) << 16);
}

// ---------------------------------------------------------------------------
// TDM: async 2D tile load (rows x 32 bf16 elems) global -> LDS, with native
// padding (64B data + 16B pad per row) to produce the LDW=40 LDS layout.
// Descriptor fields per CDNA5 ISA ch.8 (D# group0/group1).
// This toolchain exposes the 6-arg builtin:
//   (u32x4 g0, i32x8 g1, i32x4 g2, i32x4 g3, i32x8 extra, i32 cpol)
// ---------------------------------------------------------------------------
__device__ __forceinline__ void tdm_load_2d(const unsigned short* gptr,
                                            unsigned lds_off,
                                            unsigned tile_rows,
                                            unsigned tensor_rows) {
  unsigned long long ga = (unsigned long long)(size_t)gptr;
  u32x4 g0;
  g0[0] = 1u;                                    // count=1, user mode, no gather
  g0[1] = lds_off;                               // lds_addr (bytes)
  g0[2] = (unsigned)ga;                          // global_addr[95:64] -> low 32
  g0[3] = (unsigned)((ga >> 32) & 0x01FFFFFFull) // global_addr[120:96]
          | (2u << 30);                          // type=2 ("image")
  i32x8 g1;
  g1[0] = (int)((1u << 16)    // data_size = 1 -> 2 bytes
              | (1u << 20)    // pad_enable
              | (3u << 22)    // pad_interval code 3 -> every 16 DWORDs (64B)
              | (3u << 25));  // pad_amount  code 3 -> 4 DWORDs (16B)
  g1[1] = (int)((D_DIM & 0xFFFFu) << 16);        // tensor_dim0 low16 @ [63:48]
  g1[2] = (int)((tensor_rows & 0xFFFFu) << 16);  // dim0 hi=0 | tensor_dim1 low16
  g1[3] = (int)(32u << 16);                      // tile_dim0 = 32 @ [127:112]
  g1[4] = (int)tile_rows;                        // tile_dim1 @ [143:128]
  g1[5] = (int)D_DIM;                            // tensor_dim0_stride low32
  g1[6] = 0;                                     // stride hi | dim1_stride lo
  g1[7] = 0;
  i32x4 gz  = {0, 0, 0, 0};
  i32x8 gz8 = {0, 0, 0, 0, 0, 0, 0, 0};
  __builtin_amdgcn_tensor_load_to_lds(g0, g1, gz, gz, gz8, 0);
}

// ---------------------------------------------------------------------------
// 1) Split fp32 -> (bf16 hi, bf16 lo) planes, vectorized by 4
// ---------------------------------------------------------------------------
__global__ __launch_bounds__(256)
void k_split(const float* __restrict__ in, unsigned short* __restrict__ hi,
             unsigned short* __restrict__ lo, int n4) {
  int i = blockIdx.x * blockDim.x + threadIdx.x;
  if (i >= n4) return;
  float4 v = ((const float4*)in)[i];
  float f[4] = {v.x, v.y, v.z, v.w};
  unsigned h[4], l[4];
#pragma unroll
  for (int k = 0; k < 4; ++k) {
    h[k] = f2bf(f[k]);
    l[k] = f2bf(f[k] - bf2f((unsigned short)h[k]));
  }
  uint2 ph = make_uint2(h[0] | (h[1] << 16), h[2] | (h[3] << 16));
  uint2 pl = make_uint2(l[0] | (l[1] << 16), l[2] | (l[3] << 16));
  ((uint2*)hi)[i] = ph;
  ((uint2*)lo)[i] = pl;
}

// Pack [A_w(16 rows); router_w(8 rows); zeros(8 rows)] -> [32][4096] bf16 hi/lo
__global__ __launch_bounds__(256)
void k_pack_arw(const float* __restrict__ A_w, const float* __restrict__ router_w,
                unsigned short* __restrict__ hi, unsigned short* __restrict__ lo) {
  int i = blockIdx.x * blockDim.x + threadIdx.x;  // 0 .. 32*4096-1
  int row = i >> 12, col = i & 4095;
  float v = 0.0f;
  if (row < 16)      v = A_w[row * D_DIM + col];
  else if (row < 24) v = router_w[(row - 16) * D_DIM + col];
  unsigned short h = f2bf(v);
  hi[i] = h;
  lo[i] = f2bf(v - bf2f(h));
}

// Pack B_w[4096,16] -> [4096][32] bf16 hi/lo (cols 16..31 zero)
__global__ __launch_bounds__(256)
void k_pack_bw(const float* __restrict__ B_w,
               unsigned short* __restrict__ hi, unsigned short* __restrict__ lo) {
  int i = blockIdx.x * blockDim.x + threadIdx.x;  // 0 .. 4096*32-1
  int row = i >> 5, col = i & 31;
  float v = (col < 16) ? B_w[row * R_DIM + col] : 0.0f;
  unsigned short h = f2bf(v);
  hi[i] = h;
  lo[i] = f2bf(v - bf2f(h));
}

// ---------------------------------------------------------------------------
// 2) Skinny WMMA GEMM: axr[M,32] = X[M,K] @ ARW[32,K]^T  (split-bf16, 3 terms)
//    One workgroup = 8 waves, each wave does 32 rows x 32 cols.
// ---------------------------------------------------------------------------
__global__ __launch_bounds__(256)
void k_axr(const unsigned short* __restrict__ Xh, const unsigned short* __restrict__ Xl,
           const unsigned short* __restrict__ ARh, const unsigned short* __restrict__ ARl,
           float* __restrict__ axr) {
  const int lane  = threadIdx.x & 31;
  const int wid   = threadIdx.x >> 5;
  const int mbase = blockIdx.x * 256 + wid * 32;
  const int arow  = lane & 15;
  const int aoff  = (lane >> 4) * 8;   // A-frag: lanes 0-15 K[0..7]+[16..23], 16-31 K[8..15]+[24..31]
  const int boff  = (lane >> 4) * 8;   // B-frag: lanes 0-15 K[0..7]+[16..23], 16-31 K[8..15]+[24..31]

  v8f acc[2][2] = {};
  for (int kt = 0; kt < D_DIM / 32; ++kt) {
    const int kb = kt * 32;
    FragAB ah[2], al[2], bh[2], bl[2];
#pragma unroll
    for (int i = 0; i < 2; ++i) {
      const unsigned short* ph = Xh + (size_t)(mbase + i * 16 + arow) * D_DIM + kb;
      const unsigned short* pl = Xl + (size_t)(mbase + i * 16 + arow) * D_DIM + kb;
      ah[i].q[0] = *(const uint4*)(ph + aoff);
      ah[i].q[1] = *(const uint4*)(ph + 16 + aoff);
      al[i].q[0] = *(const uint4*)(pl + aoff);
      al[i].q[1] = *(const uint4*)(pl + 16 + aoff);
    }
#pragma unroll
    for (int j = 0; j < 2; ++j) {
      const unsigned short* ph = ARh + (size_t)(j * 16 + arow) * D_DIM + kb;
      const unsigned short* pl = ARl + (size_t)(j * 16 + arow) * D_DIM + kb;
      bh[j].q[0] = *(const uint4*)(ph + boff);
      bh[j].q[1] = *(const uint4*)(ph + boff + 16);
      bl[j].q[0] = *(const uint4*)(pl + boff);
      bl[j].q[1] = *(const uint4*)(pl + boff + 16);
    }
#pragma unroll
    for (int i = 0; i < 2; ++i)
#pragma unroll
      for (int j = 0; j < 2; ++j) {
        acc[i][j] = wmma_bf16(ah[i].v, bh[j].v, acc[i][j]);
        acc[i][j] = wmma_bf16(ah[i].v, bl[j].v, acc[i][j]);
        acc[i][j] = wmma_bf16(al[i].v, bh[j].v, acc[i][j]);
      }
  }
  // C layout: VGPR v -> M = v (lanes 0-15) / v+8 (lanes 16-31); N = lane & 15
#pragma unroll
  for (int i = 0; i < 2; ++i)
#pragma unroll
    for (int j = 0; j < 2; ++j) {
      const int n = j * 16 + (lane & 15);
#pragma unroll
      for (int v = 0; v < 8; ++v) {
        const int m = mbase + i * 16 + ((lane < 16) ? v : v + 8);
        axr[(size_t)m * 32 + n] = acc[i][j][v];
      }
    }
}

// ---------------------------------------------------------------------------
// 3) Per-token: softmax(router logits) + expert-core mixing -> bf16 hi/lo,
//    scaled by K_SCALING, padded to K=32 (cols 16..31 zero). 1 thread / token.
// ---------------------------------------------------------------------------
__global__ __launch_bounds__(256)
void k_mix(const float* __restrict__ axr, const float* __restrict__ cores,
           unsigned short* __restrict__ mxh, unsigned short* __restrict__ mxl) {
  __shared__ float sc[E_DIM * R_DIM * R_DIM];  // 8 KB of expert cores
  for (int i = threadIdx.x; i < E_DIM * R_DIM * R_DIM; i += blockDim.x)
    sc[i] = cores[i];
  __syncthreads();

  const int m = blockIdx.x * blockDim.x + threadIdx.x;  // 0..8191
  const float* row = axr + (size_t)m * 32;
  float ax[16], lg[8];
#pragma unroll
  for (int r = 0; r < 16; ++r) ax[r] = row[r];
#pragma unroll
  for (int e = 0; e < 8; ++e) lg[e] = row[16 + e];
  float mx = lg[0];
#pragma unroll
  for (int e = 1; e < 8; ++e) mx = fmaxf(mx, lg[e]);
  float s = 0.0f;
#pragma unroll
  for (int e = 0; e < 8; ++e) { lg[e] = __expf(lg[e] - mx); s += lg[e]; }
  const float inv = 1.0f / s;

  float acc[16] = {};
#pragma unroll
  for (int e = 0; e < 8; ++e) {
    const float pe = lg[e] * inv;
#pragma unroll
    for (int r = 0; r < 16; ++r) {
      const float a = ax[r] * pe;
      const float* c = sc + (e * 16 + r) * 16;
#pragma unroll
      for (int q = 0; q < 16; ++q) acc[q] = fmaf(a, c[q], acc[q]);
    }
  }
#pragma unroll
  for (int q = 0; q < 16; ++q) {
    const float v = acc[q] * K_SCALING;
    const unsigned short h = f2bf(v);
    mxh[(size_t)m * 32 + q] = h;
    mxl[(size_t)m * 32 + q] = f2bf(v - bf2f(h));
  }
#pragma unroll
  for (int q = 16; q < 32; ++q) {
    mxh[(size_t)m * 32 + q] = 0;
    mxl[(size_t)m * 32 + q] = 0;
  }
}

// ---------------------------------------------------------------------------
// 4) Main GEMM: out[M,N] = X @ W^T (split-bf16, 3 WMMA terms) + rank-16
//    correction (3 WMMA terms via mixed/B_w pads) + bias.
//    128x128 tile / workgroup, 8 waves of 32x64, BK=32.
//    Double-buffered LDS (2 x 4 planes x 10,240 B = 80 KB) filled by the
//    Tensor Data Mover (async DMA, TENSORcnt), overlapped with WMMA compute.
// ---------------------------------------------------------------------------
#define LDW 40  // padded row stride (halves): conflict-free b128 fragment reads

__global__ __launch_bounds__(256)
void k_gemm(const unsigned short* __restrict__ Xh, const unsigned short* __restrict__ Xl,
            const unsigned short* __restrict__ Wh, const unsigned short* __restrict__ Wl,
            const unsigned short* __restrict__ MXh, const unsigned short* __restrict__ MXl,
            const unsigned short* __restrict__ BWh, const unsigned short* __restrict__ BWl,
            const float* __restrict__ b_base, float* __restrict__ out) {
  __shared__ __align__(16) unsigned short tiles[2][4][128 * LDW];  // 81,920 B

  const int t = threadIdx.x, lane = t & 31, wid = t >> 5;
  const int mw = wid >> 1;      // 0..3  -> 32-row band
  const int nw = wid & 1;       // 0..1  -> 64-col band
  const int row0 = blockIdx.y * 128;
  const int col0 = blockIdx.x * 128;

  const unsigned short* gsrc[4] = {
      Xh + (size_t)row0 * D_DIM, Xl + (size_t)row0 * D_DIM,
      Wh + (size_t)col0 * D_DIM, Wl + (size_t)col0 * D_DIM};
  const unsigned trows[4] = {M_TOK, M_TOK, O_DIM, O_DIM};

  const int KT = D_DIM / 32;  // 128

  // prologue: DMA tile 0 into buffer 0 (wave 0 owns the TDM)
  if (wid == 0) {
#pragma unroll
    for (int a = 0; a < 4; ++a)
      tdm_load_2d(gsrc[a], (unsigned)(size_t)&tiles[0][a][0], 128, trows[a]);
  }

  v8f acc[2][4] = {};
  const int arow = lane & 15;
  const int aoff = (lane >> 4) * 8;
  const int boff = (lane >> 4) * 8;

  for (int kt = 0; kt < KT; ++kt) {
    const int buf = kt & 1;
    if (wid == 0) {
      if (kt + 1 < KT) {
        // issue next tile into the other buffer, then wait for current tile
        // (TENSORcnt retires in order: <=4 left means tile kt is complete)
#pragma unroll
        for (int a = 0; a < 4; ++a)
          tdm_load_2d(gsrc[a] + (size_t)(kt + 1) * 32,
                      (unsigned)(size_t)&tiles[buf ^ 1][a][0], 128, trows[a]);
        __builtin_amdgcn_s_wait_tensorcnt(4);
      } else {
        __builtin_amdgcn_s_wait_tensorcnt(0);
      }
    }
    __syncthreads();  // tile kt visible to all waves

    const unsigned short* sXh = &tiles[buf][0][0];
    const unsigned short* sXl = &tiles[buf][1][0];
    const unsigned short* sWh = &tiles[buf][2][0];
    const unsigned short* sWl = &tiles[buf][3][0];

    FragAB ah[2], al[2];
#pragma unroll
    for (int i = 0; i < 2; ++i) {
      const unsigned short* ph = sXh + (size_t)(mw * 32 + i * 16 + arow) * LDW;
      const unsigned short* pl = sXl + (size_t)(mw * 32 + i * 16 + arow) * LDW;
      ah[i].q[0] = *(const uint4*)(ph + aoff);
      ah[i].q[1] = *(const uint4*)(ph + 16 + aoff);
      al[i].q[0] = *(const uint4*)(pl + aoff);
      al[i].q[1] = *(const uint4*)(pl + 16 + aoff);
    }
#pragma unroll
    for (int j = 0; j < 4; ++j) {
      FragAB bh, bl;
      const unsigned short* ph = sWh + (size_t)(nw * 64 + j * 16 + arow) * LDW;
      const unsigned short* pl = sWl + (size_t)(nw * 64 + j * 16 + arow) * LDW;
      bh.q[0] = *(const uint4*)(ph + boff);
      bh.q[1] = *(const uint4*)(ph + boff + 16);
      bl.q[0] = *(const uint4*)(pl + boff);
      bl.q[1] = *(const uint4*)(pl + boff + 16);
#pragma unroll
      for (int i = 0; i < 2; ++i) {
        acc[i][j] = wmma_bf16(ah[i].v, bh.v, acc[i][j]);
        acc[i][j] = wmma_bf16(ah[i].v, bl.v, acc[i][j]);
        acc[i][j] = wmma_bf16(al[i].v, bh.v, acc[i][j]);
      }
    }
    __syncthreads();  // done reading buf before TDM overwrites it next iter
  }

  // Fused rank-16 correction epilogue: acc += mixed_pad @ bw_pad^T (3 terms)
  FragAB mh[2], ml[2];
#pragma unroll
  for (int i = 0; i < 2; ++i) {
    const unsigned short* ph = MXh + (size_t)(row0 + mw * 32 + i * 16 + arow) * 32;
    const unsigned short* pl = MXl + (size_t)(row0 + mw * 32 + i * 16 + arow) * 32;
    mh[i].q[0] = *(const uint4*)(ph + aoff);
    mh[i].q[1] = *(const uint4*)(ph + 16 + aoff);
    ml[i].q[0] = *(const uint4*)(pl + aoff);
    ml[i].q[1] = *(const uint4*)(pl + 16 + aoff);
  }
#pragma unroll
  for (int j = 0; j < 4; ++j) {
    FragAB ch, cl;
    const unsigned short* ph = BWh + (size_t)(col0 + nw * 64 + j * 16 + arow) * 32;
    const unsigned short* pl = BWl + (size_t)(col0 + nw * 64 + j * 16 + arow) * 32;
    ch.q[0] = *(const uint4*)(ph + boff);
    ch.q[1] = *(const uint4*)(ph + boff + 16);
    cl.q[0] = *(const uint4*)(pl + boff);
    cl.q[1] = *(const uint4*)(pl + boff + 16);
#pragma unroll
    for (int i = 0; i < 2; ++i) {
      acc[i][j] = wmma_bf16(mh[i].v, ch.v, acc[i][j]);
      acc[i][j] = wmma_bf16(mh[i].v, cl.v, acc[i][j]);
      acc[i][j] = wmma_bf16(ml[i].v, ch.v, acc[i][j]);
    }
  }

  // Bias + store (C layout: VGPR v -> M = v / v+8 by lane half; N = lane&15)
#pragma unroll
  for (int j = 0; j < 4; ++j) {
    const int n = col0 + nw * 64 + j * 16 + (lane & 15);
    const float bias = b_base[n];
#pragma unroll
    for (int i = 0; i < 2; ++i) {
      const int mbase2 = row0 + mw * 32 + i * 16 + ((lane < 16) ? 0 : 8);
#pragma unroll
      for (int v = 0; v < 8; ++v)
        out[(size_t)(mbase2 + v) * O_DIM + n] = acc[i][j][v] + bias;
    }
  }
}

// ---------------------------------------------------------------------------
// Launcher
// ---------------------------------------------------------------------------
extern "C" void kernel_launch(void* const* d_in, const int* in_sizes, int n_in,
                              void* d_out, int out_size, void* d_ws, size_t ws_size,
                              hipStream_t stream) {
  const float* x     = (const float*)d_in[0];  // [B,S,D]
  const float* Wb    = (const float*)d_in[1];  // [O,D]
  const float* bb    = (const float*)d_in[2];  // [O]
  const float* Aw    = (const float*)d_in[3];  // [R,D]
  const float* Bw    = (const float*)d_in[4];  // [O,R]
  const float* Rw    = (const float*)d_in[5];  // [E,D]
  const float* cores = (const float*)d_in[6];  // [E,R,R]
  float* out = (float*)d_out;

  char* ws = (char*)d_ws;
  size_t off = 0;
  auto alloc = [&](size_t bytes) -> void* {
    void* p = ws + off;
    off += (bytes + 255) & ~(size_t)255;
    return p;
  };
  unsigned short* Xh  = (unsigned short*)alloc((size_t)M_TOK * D_DIM * 2);
  unsigned short* Xl  = (unsigned short*)alloc((size_t)M_TOK * D_DIM * 2);
  unsigned short* Whp = (unsigned short*)alloc((size_t)O_DIM * D_DIM * 2);
  unsigned short* Wlp = (unsigned short*)alloc((size_t)O_DIM * D_DIM * 2);
  unsigned short* ARh = (unsigned short*)alloc((size_t)32 * D_DIM * 2);
  unsigned short* ARl = (unsigned short*)alloc((size_t)32 * D_DIM * 2);
  unsigned short* BWh = (unsigned short*)alloc((size_t)O_DIM * 32 * 2);
  unsigned short* BWl = (unsigned short*)alloc((size_t)O_DIM * 32 * 2);
  float*          axr = (float*)alloc((size_t)M_TOK * 32 * 4);
  unsigned short* MXh = (unsigned short*)alloc((size_t)M_TOK * 32 * 2);
  unsigned short* MXl = (unsigned short*)alloc((size_t)M_TOK * 32 * 2);

  // 1) precision split / packing
  {
    int n4 = M_TOK * D_DIM / 4;
    k_split<<<(n4 + 255) / 256, 256, 0, stream>>>(x, Xh, Xl, n4);
  }
  {
    int n4 = O_DIM * D_DIM / 4;
    k_split<<<(n4 + 255) / 256, 256, 0, stream>>>(Wb, Whp, Wlp, n4);
  }
  k_pack_arw<<<(32 * D_DIM + 255) / 256, 256, 0, stream>>>(Aw, Rw, ARh, ARl);
  k_pack_bw<<<(O_DIM * 32 + 255) / 256, 256, 0, stream>>>(Bw, BWh, BWl);

  // 2) ax + router logits via WMMA
  k_axr<<<M_TOK / 256, 256, 0, stream>>>(Xh, Xl, ARh, ARl, axr);

  // 3) softmax + expert-core mixing
  k_mix<<<M_TOK / 256, 256, 0, stream>>>(axr, cores, MXh, MXl);

  // 4) fused base GEMM (TDM-fed) + LoRA correction + bias
  dim3 grid(O_DIM / 128, M_TOK / 128);
  k_gemm<<<grid, 256, 0, stream>>>(Xh, Xl, Whp, Wlp, MXh, MXl, BWh, BWl, bb, out);
}
